// ElementalGTO_63668595196304
// MI455X (gfx1250) — matrix operand order repaired and checked by reading, not compiled, not run on hardware
//
#include <hip/hip_runtime.h>

typedef __attribute__((ext_vector_type(16))) _Float16 v16h;
typedef __attribute__((ext_vector_type(8)))  _Float16 v8h;
typedef __attribute__((ext_vector_type(8)))  float    v8f;

#define WAVES  4
#define BLOCK  (WAVES * 32)
#define NATOM  256
#define FPSZ   600

// W_OH weights and L index per a (a = 0..9 valid, 10..15 padding)
static constexpr float WT[16] = {1.f,1.f,1.f,1.f, 1.f,2.f,1.f,2.f,2.f,1.f, 0.f,0.f,0.f,0.f,0.f,0.f};
static constexpr int   LT[16] = {0,  1,  1,  1,   2,  2,  2,  2,  2,  2,   0,  0,  0,  0,  0,  0};

__device__ __forceinline__ void accum_partials(const float xs[8], int hi, float p[3]) {
  if (hi == 0) {
#pragma unroll
    for (int r = 0; r < 8; ++r) p[LT[r]] += WT[r] * xs[r];         // a = r
  } else {
#pragma unroll
    for (int r = 0; r < 8; ++r) p[LT[8 + r]] += WT[8 + r] * xs[r]; // a = 8 + r
  }
}

__launch_bounds__(BLOCK)
__global__ void gto_fp_wmma_kernel(const float* __restrict__ coords,
                                   const int*   __restrict__ charges,
                                   float*       __restrict__ out)
{
  __shared__ float4 atomsS[NATOM];
  __shared__ int cntS[4];                          // per-species neighbor counts
  __shared__ unsigned short listS[4][NATOM];       // species-compacted atom lists
  // per-wave WMMA staging (unmasked: species handled by K-partition)
  __shared__ __align__(128) _Float16 angS[WAVES][16][32]; // [a-row][k]
  __shared__ __align__(128) _Float16 radS[WAVES][32][32]; // [g-row][k]

  const int tid  = threadIdx.x;
  const int lane = tid & 31;
  const int wid  = tid >> 5;
  const int b    = blockIdx.x >> 6;      // 64 blocks per batch
  const int grp  = blockIdx.x & 63;
  const int n    = grp * WAVES + wid;    // atom this wave owns

  if (tid < 4) cntS[tid] = 0;
  __syncthreads();

  // ---- cache atoms + build species-compacted neighbor lists (LDS atomics) ----
  for (int t = tid; t < NATOM; t += BLOCK) {
    const float* c = coords + ((size_t)b * NATOM + t) * 3;
    int z  = charges[(size_t)b * NATOM + t];
    int sp = (z == 1) ? 0 : (z == 6) ? 1 : (z == 7) ? 2 : 3;
    atomsS[t] = make_float4(c[0], c[1], c[2], 0.0f);
    int pos = atomicAdd(&cntS[sp], 1);
    listS[sp][pos] = (unsigned short)t;
  }
  // zero pad rows once: a = 10..15 (A tile), g = 20..31 (B tiles)
#pragma unroll
  for (int a = 10; a < 16; ++a) angS[wid][a][lane] = (_Float16)0.f;
#pragma unroll
  for (int g = 20; g < 32; ++g) radS[wid][g][lane] = (_Float16)0.f;
  __syncthreads();

  const float4 cen = atomsS[n];
  const int col = lane & 15;
  const int hi  = lane >> 4;

  v8f acc[4][2];
#pragma unroll
  for (int s = 0; s < 4; ++s)
#pragma unroll
    for (int t = 0; t < 2; ++t) acc[s][t] = (v8f){0.f,0.f,0.f,0.f,0.f,0.f,0.f,0.f};

#pragma unroll
  for (int s = 0; s < 4; ++s) {
    const int c = cntS[s];                     // uniform across block
    for (int t0 = 0; t0 < c; t0 += 32) {
      const int j  = t0 + lane;
      const int jj = (j < c) ? j : (c - 1);    // clamp: branch-free ds_load
      const int m  = (int)listS[s][jj];        // duplicate tail lanes get masked
      float4 am = atomsS[m];
      float dx = cen.x - am.x, dy = cen.y - am.y, dz = cen.z - am.z;
      float d2 = dx*dx + dy*dy + dz*dz;
      bool  ok  = (j < c) && (d2 < 36.0f) && (m != n);
      float d2s = ok ? d2 : 1.0f;
      float rinv = rsqrtf(d2s);
      float d    = d2s * rinv;
      float msk  = ok ? 1.0f : 0.0f;
      float cut  = 0.5f * (__cosf(d * 0.52359878f) + 1.0f);  // pi/6

      // radial recurrence: e_g = exp(-eta*(d-mu_g)^2), mu_g = 0.3*(g+1)
      float tt   = d - 0.3f;
      float e    = __expf(-2.3f * tt * tt);     // e_0
      float v    = __expf(1.38f * tt);          // exp(0.6*eta*(d-mu_0))
      float base = 0.8556329f * cut * msk;      // sqrt(eta/pi)*fcut*mask
#pragma unroll
      for (int g = 0; g < 20; ++g) {
        radS[wid][g][lane] = (_Float16)(base * e);
        e *= v * 0.8130195f;                    // * exp(-0.09*eta)
        v *= 0.6610007f;                        // * exp(-0.18*eta)
      }

      // angular monomials: (1/d)^(2+l) * x^n y^m z^k, masked
      float r2 = rinv * rinv;
      float a0 = msk * r2;
      float s3 = a0 * rinv;
      float s4 = s3 * rinv;
      float ang[10];
      ang[0] = a0;
      ang[1] = s3*dx;   ang[2] = s3*dy;   ang[3] = s3*dz;
      ang[4] = s4*dx*dx; ang[5] = s4*dx*dy; ang[6] = s4*dy*dy;
      ang[7] = s4*dx*dz; ang[8] = s4*dy*dz; ang[9] = s4*dz*dz;
#pragma unroll
      for (int a = 0; a < 10; ++a)
        angS[wid][a][lane] = (_Float16)ang[a];

      __syncthreads();   // publish staging in WMMA layout (uniform trip count)

      // B tiles (32x16 f16): lane = column N=col, K = hi*16 .. hi*16+15
      const v16h bt0 = *(const v16h*)&radS[wid][col][hi * 16];       // g = 0..15
      const v16h bt1 = *(const v16h*)&radS[wid][16 + col][hi * 16];  // g = 16..19 (+pad)
      // A tile (16x32 f16): lane row M=col; K = hi*16+{0..7}, then hi*16+8+{0..7}
      v8h lo = *(const v8h*)&angS[wid][col][hi * 16];
      v8h hh = *(const v8h*)&angS[wid][col][hi * 16 + 8];
      v16h at = __builtin_shufflevector(lo, hh, 0,1,2,3,4,5,6,7,8,9,10,11,12,13,14,15);

      acc[s][0] = __builtin_amdgcn_wmma_f32_16x16x32_f16(false, at, false, bt0,
                                                         (short)0, acc[s][0], false, false);
      acc[s][1] = __builtin_amdgcn_wmma_f32_16x16x32_f16(false, at, false, bt1,
                                                         (short)0, acc[s][1], false, false);
      // no trailing barrier needed: per-wave staging, LDS is in-order per wave
    }
  }

  // ---- epilogue: fp[l, ch, g] = sum_a W[a]*(l==L[a]) * X, X = t_i*t_j*(i==j?1:2)
  // C/D layout: VGPR r on lane (col,hi) = test[a = hi*8 + r][g = tile*16 + col]
  const size_t base = ((size_t)b * NATOM + n) * (size_t)FPSZ;
  const int ci[10] = {0,1,2,3, 0,0,0,1,1,2};
  const int cj[10] = {0,1,2,3, 1,2,3,2,3,3};

#pragma unroll
  for (int ch = 0; ch < 10; ++ch) {
    const int   i = ci[ch], j = cj[ch];
    const float scale = (ch < 4) ? 1.0f : 2.0f;   // combos: (ti+tj)^2-ti^2-tj^2 = 2*ti*tj
    float xs0[8], xs1[8];
#pragma unroll
    for (int r = 0; r < 8; ++r) {
      xs0[r] = scale * acc[i][0][r] * acc[j][0][r];
      xs1[r] = scale * acc[i][1][r] * acc[j][1][r];
    }
    float p0[3] = {0.f,0.f,0.f}, p1[3] = {0.f,0.f,0.f};
    accum_partials(xs0, hi, p0);
    accum_partials(xs1, hi, p1);
#pragma unroll
    for (int l = 0; l < 3; ++l) {
      float f0 = p0[l] + __shfl_xor(p0[l], 16, 32);  // combine a<8 with a>=8 half
      float f1 = p1[l] + __shfl_xor(p1[l], 16, 32);
      float val = hi ? f1 : f0;
      int   g   = hi ? (16 + col) : col;             // hi lanes store the g>=16 tile
      if (hi == 0 || col < 4)
        out[base + (size_t)l * 200 + (size_t)ch * 20 + g] = val;
    }
  }
}

extern "C" void kernel_launch(void* const* d_in, const int* in_sizes, int n_in,
                              void* d_out, int out_size, void* d_ws, size_t ws_size,
                              hipStream_t stream) {
  (void)n_in; (void)out_size; (void)d_ws; (void)ws_size;
  const float* coords  = (const float*)d_in[0];
  const int*   charges = (const int*)d_in[1];
  float*       out     = (float*)d_out;
  const int B = in_sizes[0] / (NATOM * 3);   // 16
  gto_fp_wmma_kernel<<<dim3(B * 64), dim3(BLOCK), 0, stream>>>(coords, charges, out);
}
